// SoftGatingMoE_49185965473964
// MI455X (gfx1250) — compile-verified, matrix operand order checked by reading, not run on hardware
//
#include <hip/hip_runtime.h>

// ---------- problem constants ----------
#define Tt   8192     // B*S tokens
#define Dd   1024
#define Hh   1024
#define Ee   8
#define Bb   4
#define Ss   2048

// ---------- GEMM tiling ----------
#define BM 128
#define BN 64
#define BK 32

typedef __bf16 bf16x16 __attribute__((ext_vector_type(16)));
typedef __bf16 bf16x8  __attribute__((ext_vector_type(8)));
typedef float  f32x8   __attribute__((ext_vector_type(8)));
typedef int    v4i     __attribute__((ext_vector_type(4)));

// ---------- async global->LDS (gfx1250) ----------
#if defined(__has_builtin)
#if __has_builtin(__builtin_amdgcn_global_load_async_to_lds_b128)
#define HAVE_ASYNC_LDS 1
#endif
#endif
#ifndef HAVE_ASYNC_LDS
#define HAVE_ASYNC_LDS 0
#endif

#if HAVE_ASYNC_LDS
typedef __attribute__((address_space(1))) v4i* g_v4i_ptr;
typedef __attribute__((address_space(3))) v4i* l_v4i_ptr;
#if __has_builtin(__builtin_amdgcn_s_wait_asynccnt)
#define WAIT_ASYNC(n) __builtin_amdgcn_s_wait_asynccnt(n)
#else
#define WAIT_ASYNC(n) asm volatile("s_wait_asynccnt %0" ::"i"(n) : "memory")
#endif
#else
#define WAIT_ASYNC(n) ((void)0)
#endif

__device__ __forceinline__ void cp16_g2l(const __bf16* g, __bf16* l) {
#if HAVE_ASYNC_LDS
  __builtin_amdgcn_global_load_async_to_lds_b128(
      (g_v4i_ptr)(void*)(g), (l_v4i_ptr)(void*)(l), 0, 0);
#else
  *reinterpret_cast<uint4*>(l) = *reinterpret_cast<const uint4*>(g);
#endif
}

// ---------- WMMA helpers ----------
__device__ __forceinline__ f32x8 wmma_bf16(bf16x16 a, bf16x16 b, f32x8 c) {
  return __builtin_amdgcn_wmma_f32_16x16x32_bf16(
      false, a, false, b, (short)0, c, false, false);
}

// Load a 16x32 bf16 fragment from an LDS tile stored row-major [rows][BK].
// ISA layout: lanes 0-15 -> rows, K 0..7 + 16..23; lanes 16-31 -> K 8..15 + 24..31.
__device__ __forceinline__ bf16x16 ldsfrag(const __bf16* tile, int rowBase, int lane) {
  const int r    = rowBase + (lane & 15);
  const int koff = (lane >> 4) << 3;        // 0 or 8
  const __bf16* p = tile + r * BK + koff;
  bf16x8 lo = *reinterpret_cast<const bf16x8*>(p);
  bf16x8 hi = *reinterpret_cast<const bf16x8*>(p + 16);
  bf16x16 f;
#pragma unroll
  for (int i = 0; i < 8; ++i) { f[i] = lo[i]; f[i + 8] = hi[i]; }
  return f;
}

// ---------- conversion kernels ----------
__global__ __launch_bounds__(256) void cvt_x_kernel(const float* __restrict__ x,
                                                    __bf16* __restrict__ xb) {
  size_t i = ((size_t)blockIdx.x * 256 + threadIdx.x) * 4;
  float4 v = *reinterpret_cast<const float4*>(x + i);
  xb[i + 0] = (__bf16)v.x; xb[i + 1] = (__bf16)v.y;
  xb[i + 2] = (__bf16)v.z; xb[i + 3] = (__bf16)v.w;
}

// Per-expert weight convert+transpose into K-contiguous bf16 buffers:
//   w1t[h][d] = w1[e][d][h], w3t[h][d] = w3[e][d][h], w2t[d][h] = w2[e][h][d]
__global__ __launch_bounds__(256) void cvt_w_kernel(
    const float* __restrict__ w1, const float* __restrict__ w2,
    const float* __restrict__ w3, __bf16* __restrict__ w1t,
    __bf16* __restrict__ w3t, __bf16* __restrict__ w2t, int e) {
  int idx = blockIdx.x * 256 + threadIdx.x;   // 0 .. H*D-1
  int a = idx >> 10;                          // slow index of output
  int b = idx & 1023;                         // fast (contiguous) index of output
  size_t eb = (size_t)e * Dd * Hh;
  if (blockIdx.y == 0)      w1t[idx] = (__bf16)w1[eb + (size_t)b * Hh + a];
  else if (blockIdx.y == 1) w3t[idx] = (__bf16)w3[eb + (size_t)b * Hh + a];
  else                      w2t[idx] = (__bf16)w2[eb + (size_t)b * Dd + a];
}

__global__ __launch_bounds__(256) void zero_kernel(float* __restrict__ p) {
  p[blockIdx.x * 256 + threadIdx.x] = 0.f;
}

// ---------- gating: logits -> softmax -> routing + top2 combine + nonpad ----------
__global__ __launch_bounds__(256) void gate_kernel(
    const float* __restrict__ x, const float* __restrict__ gw,
    const unsigned char* __restrict__ tpad, const unsigned char* __restrict__ tmask,
    float* __restrict__ routing, float* __restrict__ combine,
    float* __restrict__ npw) {
  const int lane = threadIdx.x & 31;
  const int wave = threadIdx.x >> 5;
  const int t = blockIdx.x * 8 + wave;            // one token per wave
  const float* xr = x + (size_t)t * Dd;
  float acc[Ee] = {};
  for (int d = lane; d < Dd; d += 32) {
    float xv = xr[d];
    const float* g = gw + d * Ee;
#pragma unroll
    for (int e = 0; e < Ee; ++e) acc[e] = fmaf(xv, g[e], acc[e]);
  }
#pragma unroll
  for (int e = 0; e < Ee; ++e)
    for (int off = 16; off > 0; off >>= 1) acc[e] += __shfl_xor(acc[e], off);
  if (lane == 0) {
    float mx = acc[0];
#pragma unroll
    for (int e = 1; e < Ee; ++e) mx = fmaxf(mx, acc[e]);
    float p[Ee], sum = 0.f;
#pragma unroll
    for (int e = 0; e < Ee; ++e) { p[e] = __expf(acc[e] - mx); sum += p[e]; }
    float inv = 1.f / sum;
#pragma unroll
    for (int e = 0; e < Ee; ++e) { p[e] *= inv; routing[(size_t)t * Ee + e] = p[e]; }
    int i1 = 0;
#pragma unroll
    for (int e = 1; e < Ee; ++e) if (p[e] > p[i1]) i1 = e;
    int i2 = -1;
#pragma unroll
    for (int e = 0; e < Ee; ++e)
      if (e != i1 && (i2 < 0 || p[e] > p[i2])) i2 = e;
    float s2 = 1.f / (p[i1] + p[i2]);
#pragma unroll
    for (int e = 0; e < Ee; ++e)
      combine[(size_t)t * Ee + e] =
          (e == i1) ? p[i1] * s2 : ((e == i2) ? p[i2] * s2 : 0.f);
    npw[t] = (tpad[t] | tmask[t]) ? 0.f : 1.f;    // nonpad weight
  }
}

// ---------- GEMM1: hact = silu(x@w1) * (x@w3), bf16 out (double-buffered) ----------
__global__ __launch_bounds__(256) void gemm1_kernel(
    const __bf16* __restrict__ xb, const __bf16* __restrict__ w1t,
    const __bf16* __restrict__ w3t, __bf16* __restrict__ hact) {
  __shared__ __attribute__((aligned(16))) __bf16 As[2][BM * BK];
  __shared__ __attribute__((aligned(16))) __bf16 B1s[2][BN * BK];
  __shared__ __attribute__((aligned(16))) __bf16 B3s[2][BN * BK];

  const int tid = threadIdx.x;
  const int lane = tid & 31;
  const int wave = tid >> 5;
  const int waveM = wave & 3;   // 4 waves in M
  const int waveN = wave >> 2;  // 2 waves in N
  const int m0 = blockIdx.x * BM;
  const int n0 = blockIdx.y * BN;

  f32x8 acc1[2][2] = {}, acc3[2][2] = {};
  const int arow = tid >> 2, acol = (tid & 3) * 8;   // 16B chunk coords
  const int arow2 = (tid + 256) >> 2, acol2 = ((tid + 256) & 3) * 8;

  // prologue: stage k0 = 0 into buffer 0 (4 async ops per thread)
  cp16_g2l(xb + (size_t)(m0 + arow) * Dd + acol, &As[0][arow * BK + acol]);
  cp16_g2l(xb + (size_t)(m0 + arow2) * Dd + acol2, &As[0][arow2 * BK + acol2]);
  cp16_g2l(w1t + (size_t)(n0 + arow) * Dd + acol, &B1s[0][arow * BK + acol]);
  cp16_g2l(w3t + (size_t)(n0 + arow) * Dd + acol, &B3s[0][arow * BK + acol]);

  int cur = 0;
  for (int k0 = 0; k0 < Dd; k0 += BK) {
    const int nxt = cur ^ 1;
    if (k0 + BK < Dd) {
      const int kn = k0 + BK;
      // stage next tile while current is being consumed
      cp16_g2l(xb + (size_t)(m0 + arow) * Dd + kn + acol, &As[nxt][arow * BK + acol]);
      cp16_g2l(xb + (size_t)(m0 + arow2) * Dd + kn + acol2, &As[nxt][arow2 * BK + acol2]);
      cp16_g2l(w1t + (size_t)(n0 + arow) * Dd + kn + acol, &B1s[nxt][arow * BK + acol]);
      cp16_g2l(w3t + (size_t)(n0 + arow) * Dd + kn + acol, &B3s[nxt][arow * BK + acol]);
      if (kn + BK < Dd)  // keep L2 one more stage ahead
        __builtin_prefetch(xb + (size_t)(m0 + arow) * Dd + kn + BK + acol, 0, 1);
      WAIT_ASYNC(4);     // in-order completion: drains only the previous stage
    } else {
      WAIT_ASYNC(0);
    }
    __syncthreads();

    bf16x16 a[2], b1[2], b3[2];
#pragma unroll
    for (int mt = 0; mt < 2; ++mt) a[mt] = ldsfrag(As[cur], waveM * 32 + mt * 16, lane);
#pragma unroll
    for (int nt = 0; nt < 2; ++nt) {
      b1[nt] = ldsfrag(B1s[cur], waveN * 32 + nt * 16, lane);
      b3[nt] = ldsfrag(B3s[cur], waveN * 32 + nt * 16, lane);
    }
#pragma unroll
    for (int mt = 0; mt < 2; ++mt)
#pragma unroll
      for (int nt = 0; nt < 2; ++nt) {
        acc1[mt][nt] = wmma_bf16(a[mt], b1[nt], acc1[mt][nt]);
        acc3[mt][nt] = wmma_bf16(a[mt], b3[nt], acc3[mt][nt]);
      }
    __syncthreads();   // all waves done reading buf[cur] before it is re-staged
    cur = nxt;
  }

  // epilogue: silu(a1)*a3 -> bf16.  C layout: VGPR r = row (r + 8*(lane>=16)), col = lane&15
#pragma unroll
  for (int mt = 0; mt < 2; ++mt)
#pragma unroll
    for (int nt = 0; nt < 2; ++nt) {
      int rbase = m0 + waveM * 32 + mt * 16 + ((lane >> 4) << 3);
      int col = n0 + waveN * 32 + nt * 16 + (lane & 15);
#pragma unroll
      for (int r = 0; r < 8; ++r) {
        float a1v = acc1[mt][nt][r], a3v = acc3[mt][nt][r];
        float s = a1v / (1.f + __expf(-a1v));
        hact[(size_t)(rbase + r) * Hh + col] = (__bf16)(s * a3v);
      }
    }
}

// ---------- GEMM2: out = hact@w2; final += combine*out; classifier sums ----------
__global__ __launch_bounds__(256) void gemm2_kernel(
    const __bf16* __restrict__ hact, const __bf16* __restrict__ w2t,
    const float* __restrict__ combine, const float* __restrict__ npw,
    float* __restrict__ final_out, float* __restrict__ ebd, int e, int first) {
  __shared__ __attribute__((aligned(16))) __bf16 As[2][BM * BK];
  __shared__ __attribute__((aligned(16))) __bf16 Bs[2][BN * BK];
  __shared__ float cw[BM];
  __shared__ float nw[BM];
  __shared__ float colsum[BN];

  const int tid = threadIdx.x;
  const int lane = tid & 31;
  const int wave = tid >> 5;
  const int waveM = wave & 3;
  const int waveN = wave >> 2;
  const int m0 = blockIdx.x * BM;
  const int n0 = blockIdx.y * BN;

  if (tid < BM) {
    cw[tid] = combine[(size_t)(m0 + tid) * Ee + e];
    nw[tid] = npw[m0 + tid];
  }
  if (tid < BN) colsum[tid] = 0.f;

  f32x8 acc[2][2] = {};
  const int arow = tid >> 2, acol = (tid & 3) * 8;
  const int arow2 = (tid + 256) >> 2, acol2 = ((tid + 256) & 3) * 8;

  // prologue: stage k0 = 0 into buffer 0 (3 async ops per thread)
  cp16_g2l(hact + (size_t)(m0 + arow) * Hh + acol, &As[0][arow * BK + acol]);
  cp16_g2l(hact + (size_t)(m0 + arow2) * Hh + acol2, &As[0][arow2 * BK + acol2]);
  cp16_g2l(w2t + (size_t)(n0 + arow) * Hh + acol, &Bs[0][arow * BK + acol]);

  int cur = 0;
  for (int k0 = 0; k0 < Hh; k0 += BK) {
    const int nxt = cur ^ 1;
    if (k0 + BK < Hh) {
      const int kn = k0 + BK;
      cp16_g2l(hact + (size_t)(m0 + arow) * Hh + kn + acol, &As[nxt][arow * BK + acol]);
      cp16_g2l(hact + (size_t)(m0 + arow2) * Hh + kn + acol2, &As[nxt][arow2 * BK + acol2]);
      cp16_g2l(w2t + (size_t)(n0 + arow) * Hh + kn + acol, &Bs[nxt][arow * BK + acol]);
      if (kn + BK < Hh)
        __builtin_prefetch(hact + (size_t)(m0 + arow) * Hh + kn + BK + acol, 0, 1);
      WAIT_ASYNC(3);
    } else {
      WAIT_ASYNC(0);
    }
    __syncthreads();

    bf16x16 a[2], b[2];
#pragma unroll
    for (int mt = 0; mt < 2; ++mt) a[mt] = ldsfrag(As[cur], waveM * 32 + mt * 16, lane);
#pragma unroll
    for (int nt = 0; nt < 2; ++nt) b[nt] = ldsfrag(Bs[cur], waveN * 32 + nt * 16, lane);
#pragma unroll
    for (int mt = 0; mt < 2; ++mt)
#pragma unroll
      for (int nt = 0; nt < 2; ++nt)
        acc[mt][nt] = wmma_bf16(a[mt], b[nt], acc[mt][nt]);
    __syncthreads();
    cur = nxt;
  }

  const int batch = m0 >> 11;  // 2048 tokens per batch; BM divides S
#pragma unroll
  for (int mt = 0; mt < 2; ++mt)
#pragma unroll
    for (int nt = 0; nt < 2; ++nt) {
      int rloc_base = waveM * 32 + mt * 16 + ((lane >> 4) << 3);
      int col_l = waveN * 32 + nt * 16 + (lane & 15);
      int gcol = n0 + col_l;
      float csum = 0.f;
#pragma unroll
      for (int r = 0; r < 8; ++r) {
        int rloc = rloc_base + r;
        float v = acc[mt][nt][r] * cw[rloc];
        size_t oi = (size_t)(m0 + rloc) * Dd + gcol;
        if (first) final_out[oi] = v; else final_out[oi] += v;
        csum += v * nw[rloc];
      }
      csum += __shfl_xor(csum, 16);      // fold row-halves sharing a column
      if (lane < 16) atomicAdd(&colsum[col_l], csum);
    }
  __syncthreads();
  if (tid < BN)
    atomicAdd(&ebd[((size_t)e * Bb + batch) * Dd + n0 + tid], colsum[tid]);
}

// ---------- classifier: counts, dots, cumulative logits ----------
__global__ __launch_bounds__(256) void clf_kernel(
    const float* __restrict__ ebd, const float* __restrict__ npw,
    const float* __restrict__ cwv, const float* __restrict__ cb,
    float* __restrict__ out) {
  __shared__ float counts[Bb];
  __shared__ float de[Ee][Bb];
  const int lane = threadIdx.x & 31;
  const int wave = threadIdx.x >> 5;
  if (wave < Bb) {
    float s = 0.f;
    for (int i = lane; i < Ss; i += 32) s += npw[wave * Ss + i];
    for (int off = 16; off > 0; off >>= 1) s += __shfl_xor(s, off);
    if (lane == 0) counts[wave] = s;
  }
  __syncthreads();
  // wave w handles expert e=w, batches 0..3
#pragma unroll
  for (int b = 0; b < Bb; ++b) {
    const float* row = ebd + ((size_t)wave * Bb + b) * Dd;
    float s = 0.f;
    for (int i = lane; i < Dd; i += 32) s += row[i] * cwv[i];
    for (int off = 16; off > 0; off >>= 1) s += __shfl_xor(s, off);
    if (lane == 0) de[wave][b] = s;
  }
  __syncthreads();
  if (threadIdx.x < Bb) {
    int b = threadIdx.x;
    float cum = 0.f, bias = cb[0];
#pragma unroll
    for (int e = 0; e < Ee; ++e) {
      cum += de[e][b];
      out[e * Bb + b] = cum / counts[b] + bias;
    }
  }
}

// ---------- launch ----------
extern "C" void kernel_launch(void* const* d_in, const int* in_sizes, int n_in,
                              void* d_out, int out_size, void* d_ws, size_t ws_size,
                              hipStream_t stream) {
  const float* x      = (const float*)d_in[0];
  const unsigned char* tpad  = (const unsigned char*)d_in[1];  // bool mask
  const unsigned char* tmask = (const unsigned char*)d_in[2];  // bool mask
  // d_in[3] = top_k (always 2)
  const float* gate_w = (const float*)d_in[4];
  const float* w1     = (const float*)d_in[5];
  const float* w2     = (const float*)d_in[6];   // dict order: w1, w2, w3
  const float* w3     = (const float*)d_in[7];
  const float* clf_w  = (const float*)d_in[8];
  const float* clf_b  = (const float*)d_in[9];

  float* final_out   = (float*)d_out;                        // [T,D]
  float* logits_out  = final_out + (size_t)Tt * Dd;          // [E,B,1]
  float* routing_out = logits_out + Ee * Bb;                 // [T,E]

  unsigned char* ws = (unsigned char*)d_ws;
  __bf16* xb    = (__bf16*)(ws);
  __bf16* hact  = (__bf16*)(ws + 16777216);
  __bf16* w1t   = (__bf16*)(ws + 33554432);
  __bf16* w3t   = (__bf16*)(ws + 35651584);
  __bf16* w2t   = (__bf16*)(ws + 37748736);
  float*  comb  = (float*) (ws + 39845888);
  float*  npw   = (float*) (ws + 40108032);
  float*  ebd   = (float*) (ws + 40140800);

  cvt_x_kernel<<<(Tt * Dd) / 1024, 256, 0, stream>>>(x, xb);
  gate_kernel<<<Tt / 8, 256, 0, stream>>>(x, gate_w, tpad, tmask,
                                          routing_out, comb, npw);
  zero_kernel<<<(Ee * Bb * Dd) / 256, 256, 0, stream>>>(ebd);

  for (int e = 0; e < Ee; ++e) {
    cvt_w_kernel<<<dim3((Dd * Hh) / 256, 3), 256, 0, stream>>>(
        w1, w2, w3, w1t, w3t, w2t, e);
    gemm1_kernel<<<dim3(Tt / BM, Hh / BN), 256, 0, stream>>>(xb, w1t, w3t, hact);
    gemm2_kernel<<<dim3(Tt / BM, Dd / BN), 256, 0, stream>>>(
        hact, w2t, comb, npw, final_out, ebd, e, e == 0 ? 1 : 0);
  }
  clf_kernel<<<1, 256, 0, stream>>>(ebd, npw, clf_w, clf_b, logits_out);
}
